// LigandEncoder_66297115181623
// MI455X (gfx1250) — matrix operand order repaired and checked by reading, not compiled, not run on hardware
//
#include <hip/hip_runtime.h>
#include <hip/hip_bf16.h>

// ---------------------------------------------------------------------------
// GINE x2 + global mean pool for MI455X (gfx1250, wave32).
// Node MLPs run on v_wmma_f32_16x16x32_bf16 (fp32 accum) with LDS tiles held
// in *fragment-swizzled* order so each fragment load is 2x ds_load_b128.
// Edge phases are bandwidth/atomic bound: float4 gathers + f32 atomics.
// ---------------------------------------------------------------------------

typedef __attribute__((ext_vector_type(16))) __bf16 bf16x16;
typedef __attribute__((ext_vector_type(8)))  float  v8f;

#define NDIM 44
#define EDIM 12
#define HDIM 64
#define EMBD 128

// ---- fragment-order swizzle maps (CDNA5 ISA 7.12.2, wave32) ----------------
// A 16x32 (MxK) bf16: lane holds row m=lane&15, half=lane>>4;
//   VGPR v elem j: K = (v>>2)*16 + half*8 + (v&3)*2 + j
// inverse: (m, kl) -> lane = m + 16*((kl>>3)&1)
//                     elem = ((((kl>>4)<<2)|((kl>>1)&3))<<1) | (kl&1)
__device__ __forceinline__ int a_sw_lane(int m, int kl) {
  return m + (((kl >> 3) & 1) << 4);
}
__device__ __forceinline__ int a_sw_elem(int kl) {
  return ((((kl >> 4) << 2) | ((kl >> 1) & 3)) << 1) | (kl & 1);
}
// B 32x16 (KxN) bf16: lane holds col n=lane&15, half=lane>>4;
//   VGPR v elem j: K = (v>>2)*16 + half*8 + (v&3)*2 + j  (same as A)
// inverse: (kl, n) -> lane = n + 16*((kl>>3)&1), elem = ((kl>>4)<<3) | (kl&7)
__device__ __forceinline__ int b_sw_lane(int kl, int n) {
  return n + (((kl >> 3) & 1) << 4);
}
__device__ __forceinline__ int b_sw_elem(int kl) { return ((kl >> 4) << 3) | (kl & 7); }

__device__ __forceinline__ v8f wmma_bf16(bf16x16 a, bf16x16 b, v8f c) {
  return __builtin_amdgcn_wmma_f32_16x16x32_bf16(false, a, false, b,
                                                 (short)0, c, false, false);
}
// C/D: VGPR i -> row M = i + 8*half, col N = lane&15.

__device__ __forceinline__ bf16x16 frag(const __bf16* p) {
  return *(const bf16x16*)p;  // 32B contiguous -> 2x ds_load_b128
}

// ---------------------------------------------------------------------------
// Kernel 1: edge messages layer 1.  msg = relu(x[src] + ea @ el1_w + el1_b)
// scatter-add into agg1[dst].  1 thread = 1 edge.
// ---------------------------------------------------------------------------
__global__ __launch_bounds__(256) void edge_pass1(
    const float* __restrict__ edge_attr, const int* __restrict__ ei,
    const float* __restrict__ x, const float* __restrict__ el1_w,
    const float* __restrict__ el1_b, float* __restrict__ agg1, int n_edges) {
  __shared__ float sw[EDIM * NDIM];
  __shared__ float sb[NDIM];
  for (int i = threadIdx.x; i < EDIM * NDIM; i += 256) sw[i] = el1_w[i];
  if (threadIdx.x < NDIM) sb[threadIdx.x] = el1_b[threadIdx.x];
  __syncthreads();

  int e = blockIdx.x * 256 + threadIdx.x;
  if (e >= n_edges) return;

  int src = ei[e];
  int dst = ei[n_edges + e];
  const float4* xs4 = (const float4*)(x + (size_t)src * NDIM);
  float* ag = agg1 + (size_t)dst * NDIM;
  __builtin_prefetch(xs4, 0, 3);  // global_prefetch_b8 on the gathered row

  const float4* ep = (const float4*)(edge_attr + (size_t)e * EDIM);
  float4 e0 = ep[0], e1 = ep[1], e2 = ep[2];
  float ea[EDIM] = {e0.x, e0.y, e0.z, e0.w, e1.x, e1.y,
                    e1.z, e1.w, e2.x, e2.y, e2.z, e2.w};

#pragma unroll
  for (int q = 0; q < NDIM / 4; ++q) {
    float4 xv = xs4[q];
    float xr[4] = {xv.x, xv.y, xv.z, xv.w};
#pragma unroll
    for (int j = 0; j < 4; ++j) {
      int c = q * 4 + j;
      float acc = sb[c];
#pragma unroll
      for (int i = 0; i < EDIM; ++i) acc = fmaf(ea[i], sw[i * NDIM + c], acc);
      float m = xr[j] + acc;
      if (m > 0.0f) unsafeAtomicAdd(&ag[c], m);  // relu + skip-zero atomic
    }
  }
}

// ---------------------------------------------------------------------------
// Kernel 2: node MLP 1.  h = relu( relu((x+agg1)@W1a+b1a) @ W1b + b1b )
// One wave handles 16 nodes; 8 waves / block -> 128 nodes / block.
// ---------------------------------------------------------------------------
__global__ __launch_bounds__(256) void node_mlp1(
    const float* __restrict__ x, const float* __restrict__ agg1,
    const float* __restrict__ w1a, const float* __restrict__ b1a,
    const float* __restrict__ w1b, const float* __restrict__ b1b,
    float* __restrict__ h, int n_nodes) {
  // all tile buffers in fragment-swizzled order: [kstep][ntile][lane][16]
  __shared__ __align__(32) __bf16 sW1a[2][4][32][16];  // K padded 44->64, 8KB
  __shared__ __align__(32) __bf16 sW1b[2][4][32][16];  // 8KB
  __shared__ float sB1a[HDIM], sB1b[HDIM];
  __shared__ __align__(32) __bf16 sA[8][2][32][16];    // 16KB
  __shared__ __align__(32) __bf16 sM[8][2][32][16];    // 16KB

  int tid = threadIdx.x;
  for (int idx = tid; idx < 64 * HDIM; idx += 256) {
    int k = idx >> 6, n = idx & 63;
    float va = (k < NDIM) ? w1a[k * HDIM + n] : 0.0f;
    float vb = w1b[idx];
    int ks = k >> 5, kl = k & 31, nt = n >> 4, nn = n & 15;
    int bl = b_sw_lane(kl, nn), be = b_sw_elem(kl);
    sW1a[ks][nt][bl][be] = (__bf16)va;
    sW1b[ks][nt][bl][be] = (__bf16)vb;
  }
  if (tid < HDIM) { sB1a[tid] = b1a[tid]; sB1b[tid] = b1b[tid]; }

  int wid = tid >> 5, lane = tid & 31;
  int n16 = lane & 15, half = lane >> 4;
  int base = (blockIdx.x * 8 + wid) * 16;

  // u = x + agg1 (float4 chunks), swizzle-stored as bf16, K padded to 64
  for (int idx = lane; idx < 16 * (NDIM / 4); idx += 32) {
    int r = idx / (NDIM / 4), q = idx % (NDIM / 4);
    int node = base + r;
    float vv[4] = {0.0f, 0.0f, 0.0f, 0.0f};
    if (node < n_nodes) {
      float4 a = ((const float4*)(x + (size_t)node * NDIM))[q];
      float4 b = ((const float4*)(agg1 + (size_t)node * NDIM))[q];
      vv[0] = a.x + b.x; vv[1] = a.y + b.y;
      vv[2] = a.z + b.z; vv[3] = a.w + b.w;
    }
#pragma unroll
    for (int j = 0; j < 4; ++j) {
      int c = q * 4 + j;
      int ks = c >> 5, kl = c & 31;
      sA[wid][ks][a_sw_lane(r, kl)][a_sw_elem(kl)] = (__bf16)vv[j];
    }
  }
  for (int idx = lane; idx < 16 * (64 - NDIM); idx += 32) {  // zero pad cols
    int r = idx / (64 - NDIM), c = NDIM + idx % (64 - NDIM);
    int ks = c >> 5, kl = c & 31;
    sA[wid][ks][a_sw_lane(r, kl)][a_sw_elem(kl)] = (__bf16)0.0f;
  }
  __syncthreads();

  bf16x16 af[2];
  af[0] = frag(&sA[wid][0][lane][0]);
  af[1] = frag(&sA[wid][1][lane][0]);

#pragma unroll
  for (int nt = 0; nt < 4; ++nt) {
    v8f acc = {0.f, 0.f, 0.f, 0.f, 0.f, 0.f, 0.f, 0.f};
#pragma unroll
    for (int ks = 0; ks < 2; ++ks)
      acc = wmma_bf16(af[ks], frag(&sW1a[ks][nt][lane][0]), acc);
#pragma unroll
    for (int i = 0; i < 8; ++i) {
      int m = i + half * 8;
      int k = nt * 16 + n16;  // output col = K index of next GEMM
      float v = fmaxf(acc[i] + sB1a[k], 0.0f);
      sM[wid][k >> 5][a_sw_lane(m, k & 31)][a_sw_elem(k & 31)] = (__bf16)v;
    }
  }
  // same-wave LDS write->read: DS ops are in-order within a wave

  bf16x16 af2[2];
  af2[0] = frag(&sM[wid][0][lane][0]);
  af2[1] = frag(&sM[wid][1][lane][0]);

#pragma unroll
  for (int nt = 0; nt < 4; ++nt) {
    v8f acc = {0.f, 0.f, 0.f, 0.f, 0.f, 0.f, 0.f, 0.f};
#pragma unroll
    for (int ks = 0; ks < 2; ++ks)
      acc = wmma_bf16(af2[ks], frag(&sW1b[ks][nt][lane][0]), acc);
#pragma unroll
    for (int i = 0; i < 8; ++i) {
      int m = i + half * 8;
      int node = base + m;
      if (node < n_nodes) {
        float v = fmaxf(acc[i] + sB1b[nt * 16 + n16], 0.0f);  // outer relu
        h[(size_t)node * HDIM + nt * 16 + n16] = v;
      }
    }
  }
}

// ---------------------------------------------------------------------------
// Kernel 3: edge messages layer 2.  msg = relu(h[src] + ea @ el2_w + el2_b)
// ---------------------------------------------------------------------------
__global__ __launch_bounds__(256) void edge_pass2(
    const float* __restrict__ edge_attr, const int* __restrict__ ei,
    const float* __restrict__ h, const float* __restrict__ el2_w,
    const float* __restrict__ el2_b, float* __restrict__ agg2, int n_edges) {
  __shared__ float sw[EDIM * HDIM];
  __shared__ float sb[HDIM];
  for (int i = threadIdx.x; i < EDIM * HDIM; i += 256) sw[i] = el2_w[i];
  if (threadIdx.x < HDIM) sb[threadIdx.x] = el2_b[threadIdx.x];
  __syncthreads();

  int e = blockIdx.x * 256 + threadIdx.x;
  if (e >= n_edges) return;

  int src = ei[e];
  int dst = ei[n_edges + e];
  const float4* hs4 = (const float4*)(h + (size_t)src * HDIM);
  float* ag = agg2 + (size_t)dst * HDIM;
  __builtin_prefetch(hs4, 0, 3);

  const float4* ep = (const float4*)(edge_attr + (size_t)e * EDIM);
  float4 e0 = ep[0], e1 = ep[1], e2 = ep[2];
  float ea[EDIM] = {e0.x, e0.y, e0.z, e0.w, e1.x, e1.y,
                    e1.z, e1.w, e2.x, e2.y, e2.z, e2.w};

#pragma unroll
  for (int q = 0; q < HDIM / 4; ++q) {
    float4 hv = hs4[q];
    float hr[4] = {hv.x, hv.y, hv.z, hv.w};
#pragma unroll
    for (int j = 0; j < 4; ++j) {
      int c = q * 4 + j;
      float acc = sb[c];
#pragma unroll
      for (int i = 0; i < EDIM; ++i) acc = fmaf(ea[i], sw[i * HDIM + c], acc);
      float m = hr[j] + acc;
      if (m > 0.0f) unsafeAtomicAdd(&ag[c], m);
    }
  }
}

// ---------------------------------------------------------------------------
// Kernel 4: per-graph node counts
// ---------------------------------------------------------------------------
__global__ __launch_bounds__(256) void count_nodes(const int* __restrict__ batch,
                                                   float* __restrict__ cnt,
                                                   int n_nodes) {
  int i = blockIdx.x * 256 + threadIdx.x;
  if (i < n_nodes) unsafeAtomicAdd(&cnt[batch[i]], 1.0f);
}

// ---------------------------------------------------------------------------
// Kernel 5: node MLP 2 + pooled sum.
// y = relu((h+agg2)@W2a+b2a) @ W2b + b2b ;  sums[batch[n]] += y
// ---------------------------------------------------------------------------
__global__ __launch_bounds__(256) void node_mlp2_pool(
    const float* __restrict__ h, const float* __restrict__ agg2,
    const int* __restrict__ batch, const float* __restrict__ w2a,
    const float* __restrict__ b2a, const float* __restrict__ w2b,
    const float* __restrict__ b2b, float* __restrict__ sums, int n_nodes) {
  __shared__ __align__(32) __bf16 sW2a[2][8][32][16];  // 16KB
  __shared__ __align__(32) __bf16 sW2b[4][8][32][16];  // 32KB
  __shared__ float sB2a[EMBD], sB2b[EMBD];
  __shared__ __align__(32) __bf16 sA[8][2][32][16];    // 16KB
  __shared__ __align__(32) __bf16 sM[8][4][32][16];    // 32KB

  int tid = threadIdx.x;
  for (int idx = tid; idx < HDIM * EMBD; idx += 256) {
    int k = idx >> 7, n = idx & 127;
    int kl = k & 31, nt = n >> 4, nn = n & 15;
    sW2a[k >> 5][nt][b_sw_lane(kl, nn)][b_sw_elem(kl)] = (__bf16)w2a[idx];
  }
  for (int idx = tid; idx < EMBD * EMBD; idx += 256) {
    int k = idx >> 7, n = idx & 127;
    int kl = k & 31, nt = n >> 4, nn = n & 15;
    sW2b[k >> 5][nt][b_sw_lane(kl, nn)][b_sw_elem(kl)] = (__bf16)w2b[idx];
  }
  if (tid < EMBD) { sB2a[tid] = b2a[tid]; sB2b[tid] = b2b[tid]; }

  int wid = tid >> 5, lane = tid & 31;
  int n16 = lane & 15, half = lane >> 4;
  int base = (blockIdx.x * 8 + wid) * 16;

  // u = h + agg2 (float4 chunks), swizzle-stored bf16
  for (int idx = lane; idx < 16 * (HDIM / 4); idx += 32) {
    int r = idx >> 4, q = idx & 15;
    int node = base + r;
    float vv[4] = {0.0f, 0.0f, 0.0f, 0.0f};
    if (node < n_nodes) {
      float4 a = ((const float4*)(h + (size_t)node * HDIM))[q];
      float4 b = ((const float4*)(agg2 + (size_t)node * HDIM))[q];
      vv[0] = a.x + b.x; vv[1] = a.y + b.y;
      vv[2] = a.z + b.z; vv[3] = a.w + b.w;
    }
#pragma unroll
    for (int j = 0; j < 4; ++j) {
      int c = q * 4 + j;
      sA[wid][c >> 5][a_sw_lane(r, c & 31)][a_sw_elem(c & 31)] = (__bf16)vv[j];
    }
  }
  __syncthreads();

  bf16x16 af[2];
  af[0] = frag(&sA[wid][0][lane][0]);
  af[1] = frag(&sA[wid][1][lane][0]);

#pragma unroll
  for (int nt = 0; nt < 8; ++nt) {
    v8f acc = {0.f, 0.f, 0.f, 0.f, 0.f, 0.f, 0.f, 0.f};
#pragma unroll
    for (int ks = 0; ks < 2; ++ks)
      acc = wmma_bf16(af[ks], frag(&sW2a[ks][nt][lane][0]), acc);
#pragma unroll
    for (int i = 0; i < 8; ++i) {
      int m = i + half * 8;
      int k = nt * 16 + n16;
      float v = fmaxf(acc[i] + sB2a[k], 0.0f);
      sM[wid][k >> 5][a_sw_lane(m, k & 31)][a_sw_elem(k & 31)] = (__bf16)v;
    }
  }

  bf16x16 af2[4];
#pragma unroll
  for (int ks = 0; ks < 4; ++ks) af2[ks] = frag(&sM[wid][ks][lane][0]);

#pragma unroll
  for (int nt = 0; nt < 8; ++nt) {
    v8f acc = {0.f, 0.f, 0.f, 0.f, 0.f, 0.f, 0.f, 0.f};
#pragma unroll
    for (int ks = 0; ks < 4; ++ks)
      acc = wmma_bf16(af2[ks], frag(&sW2b[ks][nt][lane][0]), acc);
#pragma unroll
    for (int i = 0; i < 8; ++i) {
      int m = i + half * 8;
      int node = base + m;
      if (node < n_nodes) {
        float v = acc[i] + sB2b[nt * 16 + n16];  // no relu on final layer
        int g = batch[node];
        unsafeAtomicAdd(&sums[(size_t)g * EMBD + nt * 16 + n16], v);
      }
    }
  }
}

// ---------------------------------------------------------------------------
// Kernel 6: mean = sums / max(cnt, 1)
// ---------------------------------------------------------------------------
__global__ __launch_bounds__(256) void pool_div(const float* __restrict__ sums,
                                                const float* __restrict__ cnt,
                                                float* __restrict__ out,
                                                int total) {
  int i = blockIdx.x * 256 + threadIdx.x;
  if (i < total) out[i] = sums[i] / fmaxf(cnt[i >> 7], 1.0f);
}

// ---------------------------------------------------------------------------
extern "C" void kernel_launch(void* const* d_in, const int* in_sizes, int n_in,
                              void* d_out, int out_size, void* d_ws,
                              size_t ws_size, hipStream_t stream) {
  const float* x         = (const float*)d_in[0];
  const float* edge_attr = (const float*)d_in[1];
  const int*   ei        = (const int*)d_in[2];
  const int*   batch     = (const int*)d_in[3];
  const float* el1_w     = (const float*)d_in[4];
  const float* el1_b     = (const float*)d_in[5];
  const float* w1a       = (const float*)d_in[6];
  const float* b1a       = (const float*)d_in[7];
  const float* w1b       = (const float*)d_in[8];
  const float* b1b       = (const float*)d_in[9];
  const float* el2_w     = (const float*)d_in[10];
  const float* el2_b     = (const float*)d_in[11];
  const float* w2a       = (const float*)d_in[12];
  const float* b2a       = (const float*)d_in[13];
  const float* w2b       = (const float*)d_in[14];
  const float* b2b       = (const float*)d_in[15];

  const int N = in_sizes[0] / NDIM;
  const int E = in_sizes[1] / EDIM;
  const int G = out_size / EMBD;

  // Workspace carve-up (floats)
  float* agg1 = (float*)d_ws;                       // N*44
  float* hbuf = agg1 + (size_t)N * NDIM;            // N*64
  float* agg2 = hbuf + (size_t)N * HDIM;            // N*64
  float* sums = agg2 + (size_t)N * HDIM;            // G*128
  float* cnt  = sums + (size_t)G * EMBD;            // G

  // Zero the accumulation buffers (graph-capture-safe memsets)
  hipMemsetAsync(agg1, 0, (size_t)N * NDIM * sizeof(float), stream);
  hipMemsetAsync(agg2, 0, (size_t)N * HDIM * sizeof(float), stream);
  hipMemsetAsync(sums, 0, (size_t)G * EMBD * sizeof(float), stream);
  hipMemsetAsync(cnt, 0, (size_t)G * sizeof(float), stream);

  const int eb = (E + 255) / 256;
  const int nb = (N + 127) / 128;  // 8 waves x 16 nodes per block
  const int cb = (N + 255) / 256;
  const int gb = (G * EMBD + 255) / 256;

  edge_pass1<<<eb, 256, 0, stream>>>(edge_attr, ei, x, el1_w, el1_b, agg1, E);
  node_mlp1<<<nb, 256, 0, stream>>>(x, agg1, w1a, b1a, w1b, b1b, hbuf, N);
  edge_pass2<<<eb, 256, 0, stream>>>(edge_attr, ei, hbuf, el2_w, el2_b, agg2, E);
  count_nodes<<<cb, 256, 0, stream>>>(batch, cnt, N);
  node_mlp2_pool<<<nb, 256, 0, stream>>>(hbuf, agg2, batch, w2a, b2a, w2b, b2b,
                                         sums, N);
  pool_div<<<gb, 256, 0, stream>>>(sums, cnt, (float*)d_out, G * EMBD);
}
